// ClauseGAT_24867860644196
// MI455X (gfx1250) — matrix operand order repaired and checked by reading, not compiled, not run on hardware
//
#include <hip/hip_runtime.h>
#include <hip/hip_bf16.h>
#include <math.h>

// ---------------------------------------------------------------------------
// ClauseGAT on gfx1250 (MI455X).  Dense attention aggregation + all GEMMs run
// on v_wmma_f32_16x16x32_bf16 (wave32, 16x16 tile per wave, f32 accum).
// Attention logits are never materialized: pass1 = online softmax stats,
// pass2 = regenerate normalized weights in-register as WMMA A-fragments.
// ---------------------------------------------------------------------------

typedef __attribute__((ext_vector_type(16))) __bf16 v16bf;
typedef __attribute__((ext_vector_type(8)))  __bf16 v8bf;
typedef __attribute__((ext_vector_type(8)))  float  v8f;

#define NN   4096   // nodes
#define NH   4      // heads
#define DD   64     // head dim
#define FF   256    // NH*DD
#define CC   512    // clauses
#define WRD  128    // adjacency bit-words per row (NN/32)

static __device__ __forceinline__ v8f wmma_bf16(v16bf a, v16bf b, v8f c) {
  return __builtin_amdgcn_wmma_f32_16x16x32_bf16(false, a, false, b,
                                                 (short)0, c, false, false);
}

// ---- adjacency -> bitmask (64MB -> 2MB, L2 resident afterwards) -----------
__global__ void k_adj_bits(const float* __restrict__ adj,
                           unsigned* __restrict__ bits) {
  int idx = blockIdx.x * blockDim.x + threadIdx.x;   // i*WRD + w
  int i = idx >> 7;
  int w = idx & (WRD - 1);
  const float* p = adj + (size_t)i * NN + w * 32;
  unsigned word = 0u;
#pragma unroll
  for (int b = 0; b < 32; ++b) word |= (p[b] > 0.f) ? (1u << b) : 0u;
  bits[idx] = word;
}

// ---- feature embedding: [N,8] raw -> [N,64] bf16 (38 real + zero pad) -----
__global__ void k_embed(const float* __restrict__ nf, __bf16* __restrict__ xe) {
  int n = blockIdx.x * blockDim.x + threadIdx.x;
  const float* f = nf + (size_t)n * 8;
  float e[64];
#pragma unroll
  for (int t = 0; t < 64; ++t) e[t] = 0.f;
  int nt = (int)f[0]; nt = nt < 0 ? 0 : (nt > 5 ? 5 : nt);
  e[nt] = 1.f;
  e[6] = log1pf(f[1]);
  const float dv[4] = {1.f, 0.1f, 0.01f, 0.001f};  // 10000^(-k/8), k=0,2,4,6
#pragma unroll
  for (int t = 0; t < 4; ++t) { float s = f[2] * dv[t];        e[7 + 2*t]  = __sinf(s); e[8 + 2*t]  = __cosf(s); }
#pragma unroll
  for (int t = 0; t < 4; ++t) { float s = f[3] * dv[t];        e[15 + 2*t] = __sinf(s); e[16 + 2*t] = __cosf(s); }
#pragma unroll
  for (int t = 0; t < 4; ++t) { float s = f[4] * 100.f * dv[t]; e[23 + 2*t] = __sinf(s); e[24 + 2*t] = __cosf(s); }
  int ro = (int)f[5]; ro = ro < 0 ? 0 : (ro > 4 ? 4 : ro);
  e[31 + ro] = 1.f;
  e[36] = f[6];
  e[37] = f[7];
  __bf16* out = xe + (size_t)n * 64;
#pragma unroll
  for (int t = 0; t < 64; ++t) out[t] = (__bf16)e[t];
}

// ---- weight [K,Nc] f32 -> WT [Nc,Kpad] bf16 (transposed, K zero-padded) ---
__global__ void k_w_transpose_pad(const float* __restrict__ Wsrc,
                                  __bf16* __restrict__ WT,
                                  int Krows, int Ncols, int Kpad) {
  int idx = blockIdx.x * blockDim.x + threadIdx.x;  // n*Kpad + k
  int nn = idx / Kpad;
  int k  = idx - nn * Kpad;
  if (nn >= Ncols) return;
  WT[idx] = (k < Krows) ? (__bf16)Wsrc[(size_t)k * Ncols + nn] : (__bf16)0.f;
}

__global__ void k_cvt_bf16(const float* __restrict__ in,
                           __bf16* __restrict__ out, int count) {
  int i = blockIdx.x * blockDim.x + threadIdx.x;
  if (i < count) out[i] = (__bf16)in[i];
}

// ---- h [N,FF] f32 -> hT [FF,N] bf16 (B-operand layout for attention) ------
__global__ void k_transpose_cvt(const float* __restrict__ h,
                                __bf16* __restrict__ hT) {
  int idx = blockIdx.x * blockDim.x + threadIdx.x;  // c*NN + n
  int c = idx >> 12;
  int n = idx & (NN - 1);
  hT[idx] = (__bf16)h[(size_t)n * FF + c];
}

// ---- generic WMMA GEMM: C[M,Nc] = A[M,K](bf16,row) @ BT[Nc,K](bf16,row)^T -
__global__ void k_gemm_bf16(const __bf16* __restrict__ A,
                            const __bf16* __restrict__ BT,
                            float* __restrict__ Cc, int M, int Nc, int K) {
  const int lane = threadIdx.x & 31;
  const int half = lane >> 4;
  const int r    = lane & 15;
  const int m0   = blockIdx.x * 16;
  const int n0   = blockIdx.y * 16;
  const __bf16* arow = A  + (size_t)(m0 + r) * K;
  const __bf16* brow = BT + (size_t)(n0 + r) * K;
  v8f acc = {};
  for (int k0 = 0; k0 < K; k0 += 32) {
    // A-frag: lanes 0-15 hold K 0-7 / 16-23, lanes 16-31 hold K 8-15 / 24-31
    v8bf alo = *(const v8bf*)(arow + k0 + half * 8);
    v8bf ahi = *(const v8bf*)(arow + k0 + 16 + half * 8);
    v16bf a;
#pragma unroll
    for (int t = 0; t < 8; ++t) { a[t] = alo[t]; a[t + 8] = ahi[t]; }
    // B-frag: lane = column, same K layout as A-frag (0-7/16-23 | 8-15/24-31)
    v8bf blo = *(const v8bf*)(brow + k0 + half * 8);
    v8bf bhi = *(const v8bf*)(brow + k0 + 16 + half * 8);
    v16bf b;
#pragma unroll
    for (int t = 0; t < 8; ++t) { b[t] = blo[t]; b[t + 8] = bhi[t]; }
    acc = wmma_bf16(a, b, acc);
  }
#pragma unroll
  for (int v = 0; v < 8; ++v)
    Cc[(size_t)(m0 + v + half * 8) * Nc + n0 + r] = acc[v];
}

// ---- src/dst attention projections: src[n,h] = <h[n,h,:], a_src[h,:]> -----
__global__ void k_src_dst(const float* __restrict__ h,
                          const float* __restrict__ asrc,
                          const float* __restrict__ adst,
                          float* __restrict__ src, float* __restrict__ dst) {
  int idx = blockIdx.x * blockDim.x + threadIdx.x;  // n*NH + hd
  int n  = idx >> 2;
  int hd = idx & 3;
  const float* hr = h + (size_t)n * FF + hd * DD;
  const float* as = asrc + hd * DD;
  const float* ad = adst + hd * DD;
  float a = 0.f, b = 0.f;
#pragma unroll 8
  for (int d = 0; d < DD; ++d) { float v = hr[d]; a += v * as[d]; b += v * ad[d]; }
  src[idx] = a;
  dst[idx] = b;
}

// ---- pass 1: per-row online softmax stats (max, 1/sum) over neighbors -----
__global__ void k_attn_stats(const float* __restrict__ src,
                             const float* __restrict__ dst,
                             const unsigned* __restrict__ bits,
                             float* __restrict__ rmax,
                             float* __restrict__ rinv) {
  const int wave = threadIdx.x >> 5;
  const int lane = threadIdx.x & 31;
  const int i    = blockIdx.x * 4 + wave;
  float si[NH], m[NH], s[NH];
#pragma unroll
  for (int h = 0; h < NH; ++h) { si[h] = src[i * NH + h]; m[h] = -1e30f; s[h] = 0.f; }
  const unsigned* wrow = bits + (size_t)i * WRD;
  for (int wi = lane; wi < WRD; wi += 32) {
    unsigned w = wrow[wi];
    while (w) {
      int b = __builtin_ctz(w);
      w &= w - 1;
      int j = wi * 32 + b;
#pragma unroll
      for (int h = 0; h < NH; ++h) {
        float e = si[h] + dst[j * NH + h];
        e = e > 0.f ? e : 0.2f * e;                 // leaky_relu(0.2)
        float nm = fmaxf(m[h], e);
        s[h] = s[h] * __expf(m[h] - nm) + __expf(e - nm);
        m[h] = nm;
      }
    }
  }
#pragma unroll
  for (int h = 0; h < NH; ++h) {
#pragma unroll
    for (int off = 16; off > 0; off >>= 1) {
      float om = __shfl_xor(m[h], off, 32);
      float os = __shfl_xor(s[h], off, 32);
      float nm = fmaxf(m[h], om);
      s[h] = s[h] * __expf(m[h] - nm) + os * __expf(om - nm);
      m[h] = nm;
    }
  }
  if (lane == 0) {
#pragma unroll
    for (int h = 0; h < NH; ++h) {
      rmax[i * NH + h] = m[h];
      rinv[i * NH + h] = 1.f / s[h];
    }
  }
}

// ---- pass 2: fused softmax-weight generation + WMMA aggregation -----------
// block = 16-row i-tile, 4 waves = 4 heads.  Per wave: 128 K-steps of 32 j's,
// A-frag built in-register, 4 accumulators cover d = 0..63.
__global__ void k_attn_aggregate(const __bf16* __restrict__ hT,   // [NH][DD][NN]
                                 const float* __restrict__ src,
                                 const float* __restrict__ dst,
                                 const float* __restrict__ rmax,
                                 const float* __restrict__ rinv,
                                 const unsigned* __restrict__ bits,
                                 float* __restrict__ o) {         // [NN][FF]
  const int head = threadIdx.x >> 5;
  const int lane = threadIdx.x & 31;
  const int half = lane >> 4;
  const int r    = lane & 15;
  const int i0   = blockIdx.x * 16;
  const int row  = i0 + r;                       // A-frag row for this lane
  const float s_i = src[row * NH + head];
  const float m_i = rmax[row * NH + head];
  const float v_i = rinv[row * NH + head];
  const unsigned* wrow = bits + (size_t)row * WRD;
  const __bf16* hbase  = hT + (size_t)head * DD * NN;
  v8f acc0 = {}, acc1 = {}, acc2 = {}, acc3 = {};
  for (int j0 = 0; j0 < NN; j0 += 32) {
    unsigned w = wrow[j0 >> 5];                  // 32-aligned j-tile = 1 word
    v16bf a;
#pragma unroll
    for (int t = 0; t < 16; ++t) {
      int off = t + half * 16;                   // A-frag K layout (contiguous)
      int j = j0 + off;
      float e = s_i + dst[j * NH + head];
      e = e > 0.f ? e : 0.2f * e;
      float av = ((w >> off) & 1u) ? __expf(e - m_i) * v_i : 0.f;
      a[t] = (__bf16)av;
    }
    int kb = j0 + half * 16;
    v16bf b0 = *(const v16bf*)(hbase + (size_t)(0 * 16 + r) * NN + kb);
    v16bf b1 = *(const v16bf*)(hbase + (size_t)(1 * 16 + r) * NN + kb);
    v16bf b2 = *(const v16bf*)(hbase + (size_t)(2 * 16 + r) * NN + kb);
    v16bf b3 = *(const v16bf*)(hbase + (size_t)(3 * 16 + r) * NN + kb);
    acc0 = wmma_bf16(a, b0, acc0);
    acc1 = wmma_bf16(a, b1, acc1);
    acc2 = wmma_bf16(a, b2, acc2);
    acc3 = wmma_bf16(a, b3, acc3);
  }
#pragma unroll
  for (int v = 0; v < 8; ++v) {
    int mrow = i0 + v + half * 8;                // C-frag row layout
    float* op = o + (size_t)mrow * FF + head * DD;
    op[0 * 16 + r] = acc0[v];
    op[1 * 16 + r] = acc1[v];
    op[2 * 16 + r] = acc2[v];
    op[3 * 16 + r] = acc3[v];
  }
}

// ---- layernorm(256) + ELU -> bf16 (layer-1 concat output) -----------------
__global__ void k_ln_elu_cat(const float* __restrict__ o,
                             const float* __restrict__ g,
                             const float* __restrict__ bb,
                             __bf16* __restrict__ xb) {
  const int wave = threadIdx.x >> 5;
  const int lane = threadIdx.x & 31;
  const int n    = blockIdx.x * 8 + wave;
  const float* row = o + (size_t)n * FF;
  float v[8], sum = 0.f;
#pragma unroll
  for (int t = 0; t < 8; ++t) { v[t] = row[lane + 32 * t]; sum += v[t]; }
#pragma unroll
  for (int off = 16; off > 0; off >>= 1) sum += __shfl_xor(sum, off, 32);
  float mu = sum * (1.f / FF);
  float sq = 0.f;
#pragma unroll
  for (int t = 0; t < 8; ++t) { float d = v[t] - mu; sq += d * d; }
#pragma unroll
  for (int off = 16; off > 0; off >>= 1) sq += __shfl_xor(sq, off, 32);
  float rstd = rsqrtf(sq * (1.f / FF) + 1e-5f);
#pragma unroll
  for (int t = 0; t < 8; ++t) {
    int c = lane + 32 * t;
    float y = (v[t] - mu) * rstd * g[c] + bb[c];
    y = y > 0.f ? y : (__expf(y) - 1.f);         // ELU
    xb[(size_t)n * FF + c] = (__bf16)y;
  }
}

// ---- head-mean + layernorm(64) + ELU -> x2T [DD][NN] bf16 (pool B-operand)-
__global__ void k_headmean_ln_elu(const float* __restrict__ o,
                                  const float* __restrict__ g,
                                  const float* __restrict__ bb,
                                  __bf16* __restrict__ x2T) {
  const int lane = threadIdx.x;
  const int n    = blockIdx.x;
  const float* row = o + (size_t)n * FF;
  float v0 = 0.25f * (row[lane]      + row[64 + lane] + row[128 + lane] + row[192 + lane]);
  float v1 = 0.25f * (row[32 + lane] + row[96 + lane] + row[160 + lane] + row[224 + lane]);
  float sum = v0 + v1;
#pragma unroll
  for (int off = 16; off > 0; off >>= 1) sum += __shfl_xor(sum, off, 32);
  float mu = sum * (1.f / DD);
  float sq = (v0 - mu) * (v0 - mu) + (v1 - mu) * (v1 - mu);
#pragma unroll
  for (int off = 16; off > 0; off >>= 1) sq += __shfl_xor(sq, off, 32);
  float rstd = rsqrtf(sq * (1.f / DD) + 1e-5f);
  float y0 = (v0 - mu) * rstd * g[lane] + bb[lane];
  float y1 = (v1 - mu) * rstd * g[lane + 32] + bb[lane + 32];
  y0 = y0 > 0.f ? y0 : (__expf(y0) - 1.f);
  y1 = y1 > 0.f ? y1 : (__expf(y1) - 1.f);
  x2T[(size_t)lane * NN + n]        = (__bf16)y0;
  x2T[(size_t)(lane + 32) * NN + n] = (__bf16)y1;
}

// ---- clause scorer: out[c] = relu(ce @ Ws + bs) @ Wo + bo -----------------
__global__ void k_score(const float* __restrict__ ce, const float* __restrict__ Ws,
                        const float* __restrict__ bs, const float* __restrict__ Wo,
                        const float* __restrict__ bo, float* __restrict__ out) {
  __shared__ float sh[DD];
  int c = blockIdx.x;
  int d = threadIdx.x;
  const float* cr = ce + (size_t)c * DD;
  float s = bs[d];
#pragma unroll 8
  for (int k = 0; k < DD; ++k) s += cr[k] * Ws[k * DD + d];
  s = s > 0.f ? s : 0.f;
  sh[d] = s * Wo[d];
  __syncthreads();
  if (d == 0) {
    float t = bo[0];
    for (int k = 0; k < DD; ++k) t += sh[k];
    out[c] = t;
  }
}

// ---------------------------------------------------------------------------
extern "C" void kernel_launch(void* const* d_in, const int* in_sizes, int n_in,
                              void* d_out, int out_size, void* d_ws, size_t ws_size,
                              hipStream_t stream) {
  (void)in_sizes; (void)n_in; (void)out_size; (void)ws_size;
  const float* nf   = (const float*)d_in[0];
  const float* adj  = (const float*)d_in[1];
  const float* pool = (const float*)d_in[2];
  const float* W1   = (const float*)d_in[3];
  const float* as1  = (const float*)d_in[4];
  const float* ad1  = (const float*)d_in[5];
  const float* g1   = (const float*)d_in[6];
  const float* b1   = (const float*)d_in[7];
  const float* W2   = (const float*)d_in[8];
  const float* as2  = (const float*)d_in[9];
  const float* ad2  = (const float*)d_in[10];
  const float* g2   = (const float*)d_in[11];
  const float* b2   = (const float*)d_in[12];
  const float* Ws   = (const float*)d_in[13];
  const float* bs   = (const float*)d_in[14];
  const float* Wo   = (const float*)d_in[15];
  const float* bo   = (const float*)d_in[16];
  float* out = (float*)d_out;

  char* w = (char*)d_ws;
  size_t off = 0;
  auto alloc = [&](size_t bytes) {
    void* p = w + off;
    off = (off + bytes + 255) & ~(size_t)255;
    return p;
  };
  unsigned* bits = (unsigned*)alloc((size_t)NN * WRD * 4);   // 2 MB
  __bf16* xeb  = (__bf16*)alloc((size_t)NN * 64 * 2);
  __bf16* W1T  = (__bf16*)alloc((size_t)256 * 64 * 2);
  __bf16* W2T  = (__bf16*)alloc((size_t)256 * 256 * 2);
  __bf16* PB   = (__bf16*)alloc((size_t)CC * NN * 2);
  float*  h1   = (float*) alloc((size_t)NN * FF * 4);
  __bf16* hT1  = (__bf16*)alloc((size_t)FF * NN * 2);
  float*  src1 = (float*) alloc((size_t)NN * NH * 4);
  float*  dst1 = (float*) alloc((size_t)NN * NH * 4);
  float*  m1   = (float*) alloc((size_t)NN * NH * 4);
  float*  is1  = (float*) alloc((size_t)NN * NH * 4);
  float*  o1   = (float*) alloc((size_t)NN * FF * 4);
  __bf16* x1b  = (__bf16*)alloc((size_t)NN * FF * 2);
  float*  h2   = (float*) alloc((size_t)NN * FF * 4);
  __bf16* hT2  = (__bf16*)alloc((size_t)FF * NN * 2);
  float*  src2 = (float*) alloc((size_t)NN * NH * 4);
  float*  dst2 = (float*) alloc((size_t)NN * NH * 4);
  float*  m2   = (float*) alloc((size_t)NN * NH * 4);
  float*  is2  = (float*) alloc((size_t)NN * NH * 4);
  float*  o2   = (float*) alloc((size_t)NN * FF * 4);
  __bf16* x2T  = (__bf16*)alloc((size_t)DD * NN * 2);
  float*  ce   = (float*) alloc((size_t)CC * DD * 4);

  // prep
  k_adj_bits       <<<(NN * WRD) / 256, 256, 0, stream>>>(adj, bits);
  k_embed          <<<NN / 256, 256, 0, stream>>>(nf, xeb);
  k_w_transpose_pad<<<(256 * 64) / 256, 256, 0, stream>>>(W1, W1T, 38, 256, 64);
  k_w_transpose_pad<<<(256 * 256) / 256, 256, 0, stream>>>(W2, W2T, 256, 256, 256);
  k_cvt_bf16       <<<(CC * NN) / 256, 256, 0, stream>>>(pool, PB, CC * NN);

  // ---- GAT layer 1 (concat heads) ----
  k_gemm_bf16      <<<dim3(NN / 16, FF / 16), 32, 0, stream>>>(xeb, W1T, h1, NN, FF, 64);
  k_src_dst        <<<(NN * NH) / 256, 256, 0, stream>>>(h1, as1, ad1, src1, dst1);
  k_attn_stats     <<<NN / 4, 128, 0, stream>>>(src1, dst1, bits, m1, is1);
  k_transpose_cvt  <<<(NN * FF) / 256, 256, 0, stream>>>(h1, hT1);
  k_attn_aggregate <<<NN / 16, 128, 0, stream>>>(hT1, src1, dst1, m1, is1, bits, o1);
  k_ln_elu_cat     <<<NN / 8, 256, 0, stream>>>(o1, g1, b1, x1b);

  // ---- GAT layer 2 (mean heads) ----
  k_gemm_bf16      <<<dim3(NN / 16, FF / 16), 32, 0, stream>>>(x1b, W2T, h2, NN, FF, FF);
  k_src_dst        <<<(NN * NH) / 256, 256, 0, stream>>>(h2, as2, ad2, src2, dst2);
  k_attn_stats     <<<NN / 4, 128, 0, stream>>>(src2, dst2, bits, m2, is2);
  k_transpose_cvt  <<<(NN * FF) / 256, 256, 0, stream>>>(h2, hT2);
  k_attn_aggregate <<<NN / 16, 128, 0, stream>>>(hT2, src2, dst2, m2, is2, bits, o2);
  k_headmean_ln_elu<<<NN, 32, 0, stream>>>(o2, g2, b2, x2T);

  // ---- pooling + scorer ----
  k_gemm_bf16      <<<dim3(CC / 16, DD / 16), 32, 0, stream>>>(PB, x2T, ce, CC, DD, NN);
  k_score          <<<CC, DD, 0, stream>>>(ce, Ws, bs, Wo, bo, out);
}
